// Drifter_64235530879482
// MI455X (gfx1250) — compile-verified, matrix operand order checked by reading, not run on hardware
//
#include <hip/hip_runtime.h>
#include <hip/hip_bf16.h>

// ---------------------------------------------------------------------------
// MI455X (gfx1250) implementation.
//
// Output = [t_mesh (B*T f32), xt (B*T f32)], B=1048576, T=101  => 847 MB.
// HBM roofline: 847MB / 23.3 TB/s ~= 36us. Compute is collapsed to
// 2 transcendentals + ~30 FMA per particle-step via Chebyshev recurrence,
// so the kernel is store-bandwidth bound. LDS-staged transpose gives
// contiguous 64B row segments; drained via global_store_async_from_lds_b128
// (CDNA5 async path). t_mesh tiles are produced by
// v_wmma_f32_16x16x32_f16 (exact: all values are f16-representable ints*20).
// ---------------------------------------------------------------------------

typedef __attribute__((ext_vector_type(16))) _Float16 v16h;
typedef __attribute__((ext_vector_type(8)))  float    v8f;
typedef __attribute__((ext_vector_type(4)))  int      v4i;

#define PI_F         3.14159265358979323846f
#define TWO_PI_F     6.28318530717958647692f
#define INV_TWO_PI_F 0.15915494309189533577f
#define T_PTS        101
#define DT_F         20.0f

#if defined(__gfx1250__) && \
    __has_builtin(__builtin_amdgcn_global_store_async_from_lds_b128) && \
    __has_builtin(__builtin_amdgcn_s_wait_asynccnt)
#define HAVE_ASYNC_LDS 1
#pragma message("CDNA5 probe: async-from-LDS path enabled")
// Builtin signature (from clang diagnostic): param0 = v4i in AS(1) (__device__),
// param1 = v4i in AS(3) (__shared__), then imm offset, imm cpol.
typedef __attribute__((address_space(1))) v4i gv4i;   // global int4
typedef __attribute__((address_space(3))) v4i lv4i;   // LDS int4
#else
#pragma message("CDNA5 probe: async-from-LDS builtins NOT found in this pass")
#endif

// ---------------------------------------------------------------------------
// t_mesh writer: each wave builds a 16x16 tile D[m][n] = (t0+n)*20 with one
// WMMA (A = e0 column of ones, B row0 = t values; exact in f16*f32-acc),
// then stores rows. One block = 8 waves = 8 row-tiles of 16 particles.
// ---------------------------------------------------------------------------
__global__ __launch_bounds__(256) void tmesh_wmma_kernel(float* __restrict__ out) {
    const int lane = threadIdx.x & 31;
    const int wave = threadIdx.x >> 5;
    const long rowBase = ((long)blockIdx.x * 8 + wave) * 16;

    v16h a = {};
    if (lane < 16) a[0] = (_Float16)1.0f;   // A[m][0] = 1 for all m

    for (int ct = 0; ct < 7; ++ct) {
        const int t0 = ct * 16;
        v16h bm = {};
        if (lane < 16) bm[0] = (_Float16)(float)((t0 + lane) * 20); // B[0][n]
        v8f c = {};
        // D[m][n] = sum_k A[m][k]*B[k][n] = (t0+n)*20   (f32 accumulate, exact)
        v8f d = __builtin_amdgcn_wmma_f32_16x16x32_f16(
            /*neg_a=*/false, a, /*neg_b=*/false, bm,
            /*c_mod=*/(short)0, c, /*reuse_a=*/false, /*reuse_b=*/false);

        const int n   = lane & 15;
        const int col = t0 + n;
        const long row0 = rowBase + ((lane >> 4) << 3); // lanes16-31 hold M=8..15
        if (col < T_PTS) {
#pragma unroll
            for (int r = 0; r < 8; ++r)
                out[(row0 + r) * T_PTS + col] = d[r];
        }
    }
}

// ---------------------------------------------------------------------------
// Integrator: 1 thread = 1 particle; Chebyshev recurrence for sin(kx)/cos(kx)
// (2 transcendentals per step). 16-step chunks staged in LDS (row stride 20
// floats -> 16B aligned float4 slots), drained with async-from-LDS b128
// stores (contiguous 64B per particle-row segment).
// ---------------------------------------------------------------------------
__global__ __launch_bounds__(256) void drifter_integrate_kernel(
    const float* __restrict__ x0, const float* __restrict__ sw,
    const float* __restrict__ cw, float* __restrict__ xt) {

    __shared__ __align__(16) float tile[256 * 20];

    const int  tid = threadIdx.x;
    const long b0  = (long)blockIdx.x * 256;

    float swr[8], cwr[8];
#pragma unroll
    for (int k = 0; k < 8; ++k) { swr[k] = sw[k]; cwr[k] = cw[k]; }

    float x = x0[b0 + tid];
    int tglob = 0;

    for (int chunk = 0; chunk < 7; ++chunk) {
        const int n = (chunk < 6) ? 16 : 5;   // 6*16 + 5 = 101

        for (int i = 0; i < n; ++i) {
            // wrap to [-pi, pi):  (x+pi) mod 2pi - pi  (floor-based, Python %)
            float y = x + PI_F;
            y = y - TWO_PI_F * floorf(y * INV_TWO_PI_F);
            tile[tid * 20 + i] = y - PI_F;

            if (tglob < 100) {
                const float s1 = __sinf(x);
                const float c1 = __cosf(x);
                float acc = fmaf(s1, swr[1], fmaf(c1, cwr[1], cwr[0]));
                const float t2 = c1 + c1;
                float skm2 = 0.0f, ckm2 = 1.0f, skm1 = s1, ckm1 = c1;
#pragma unroll
                for (int k = 2; k < 8; ++k) {
                    const float sk = fmaf(t2, skm1, -skm2);  // sin(kx)
                    const float ck = fmaf(t2, ckm1, -ckm2);  // cos(kx)
                    acc = fmaf(sk, swr[k], acc);
                    acc = fmaf(ck, cwr[k], acc);
                    skm2 = skm1; skm1 = sk; ckm2 = ckm1; ckm1 = ck;
                }
                x = fmaf(acc, DT_F, x);   // Euler step, dt = 20
            }
            ++tglob;
        }
        __syncthreads();

        if (chunk < 6) {
            const int t0 = chunk * 16;
            // 256 rows x 16 cols = 1024 float4s; 4 per thread.
#pragma unroll
            for (int it = 0; it < 4; ++it) {
                const int f   = tid + (it << 8);   // 0..1023
                const int row = f >> 2;            // particle within block
                const int qc  = (f & 3) << 2;      // col quad: 0,4,8,12
                const long gidx = (b0 + row) * T_PTS + t0 + qc;
#ifdef HAVE_ASYNC_LDS
                __builtin_amdgcn_global_store_async_from_lds_b128(
                    (gv4i*)(xt + gidx), (lv4i*)&tile[row * 20 + qc],
                    /*offset=*/0, /*cpol=*/0);
#else
                const float4 v = *(const float4*)&tile[row * 20 + qc];
                *(float4*)(xt + gidx) = v;
#endif
            }
#ifdef HAVE_ASYNC_LDS
            __builtin_amdgcn_s_wait_asynccnt(0); // LDS reads done before reuse
#endif
        } else {
            // tail: cols 96..100 (5 per row), 1280 scalars.
#pragma unroll
            for (int it = 0; it < 5; ++it) {
                const int f   = tid + (it << 8);   // 0..1279
                const int row = f / 5;
                const int col = f - row * 5;
                xt[(b0 + row) * T_PTS + 96 + col] = tile[row * 20 + col];
            }
        }
        __syncthreads();
    }
}

extern "C" void kernel_launch(void* const* d_in, const int* in_sizes, int n_in,
                              void* d_out, int out_size, void* d_ws, size_t ws_size,
                              hipStream_t stream) {
    const float* x0 = (const float*)d_in[0];
    const float* sw = (const float*)d_in[1];
    const float* cw = (const float*)d_in[2];
    // d_in[3] (t_sample) intentionally unused: it does not affect outputs.

    const long B  = in_sizes[0];            // 1048576
    float* out    = (float*)d_out;          // [t_mesh | xt]
    const long BT = B * (long)T_PTS;

    // t_mesh: 8 waves/block, each wave covers 16 rows x 101 cols -> B/128 blocks
    tmesh_wmma_kernel<<<(int)(B / 128), 256, 0, stream>>>(out);

    // xt: 256 particles per block
    drifter_integrate_kernel<<<(int)(B / 256), 256, 0, stream>>>(x0, sw, cw, out + BT);
}
